// GAT_42992622633737
// MI455X (gfx1250) — matrix operand order repaired and checked by reading, not compiled, not run on hardware
//
#include <hip/hip_runtime.h>
#include <hip/hip_bf16.h>

// ---------- types for WMMA ----------
typedef __bf16 v16bf __attribute__((ext_vector_type(16)));
typedef float  v8f   __attribute__((ext_vector_type(8)));

union FragBF { v16bf bf; unsigned int u[8]; };

__device__ __forceinline__ unsigned short f2bf(float f) {
  unsigned int u = __float_as_uint(f);
  u += 0x7FFFu + ((u >> 16) & 1u);          // round-to-nearest-even
  return (unsigned short)(u >> 16);
}

__device__ __forceinline__ void atomicMaxFloat(float* addr, float v) {
  // order-preserving int trick; storage initialized to -inf
  if (v >= 0.f) atomicMax((int*)addr, __float_as_int(v));
  else          atomicMin((unsigned int*)addr, __float_as_uint(v));
}

#define STR1 136   // padded halfword stride (128 + 8) -> 4-bank row skew

// ============================================================
// GEMM1: feat1[N,128] = h[N,256] @ W1[256,128]   (bf16 WMMA, f32 accum)
// block: 256 thr = 8 waves; tile 64 rows x 128 cols; K chunked 2x128
// ============================================================
__global__ __launch_bounds__(256) void gat_gemm1(const float* __restrict__ h,
                                                 const float* __restrict__ W1,
                                                 float* __restrict__ feat1, int n) {
  __shared__ unsigned short lA[64 * STR1];   // rows x k   (17.4 KB)
  __shared__ unsigned short lW[128 * STR1];  // cols x k   (34.8 KB)
  const int t    = threadIdx.x;
  const int row0 = blockIdx.x * 64;
  const int wave = t >> 5;
  const int lane = t & 31;
  const int lh   = lane >> 4;
  const int l16  = lane & 15;

  v8f z = {0.f,0.f,0.f,0.f,0.f,0.f,0.f,0.f};
  v8f c[4];
  for (int r = 0; r < 4; ++r) c[r] = z;

  for (int kc = 0; kc < 2; ++kc) {
    // A tile: 64 rows x 128 k, fp32 -> bf16 (coalesced reads)
    for (int i = 0; i < 32; ++i) {
      int lin = i * 256 + t;
      int row = lin >> 7, k = lin & 127;
      int node = row0 + row;
      float v = (node < n) ? h[(size_t)node * 256 + kc * 128 + k] : 0.f;
      lA[row * STR1 + k] = f2bf(v);
    }
    // W tile stored transposed: lW[col][k]
    for (int i = 0; i < 64; ++i) {
      int lin = i * 256 + t;
      int k = lin >> 7, col = lin & 127;
      lW[col * STR1 + k] = f2bf(W1[(size_t)(kc * 128 + k) * 128 + col]);
    }
    __syncthreads();

    #pragma unroll
    for (int kk = 0; kk < 4; ++kk) {
      // B fragment: N = l16; i<8 -> K = kk*32+lh*8+i; i>=8 -> K = kk*32+16+lh*8+(i-8)
      FragBF b;
      const unsigned int* wp =
          (const unsigned int*)&lW[(wave * 16 + l16) * STR1 + kk * 32 + lh * 8];
      #pragma unroll
      for (int j = 0; j < 4; ++j) { b.u[j] = wp[j]; b.u[4 + j] = wp[8 + j]; }

      #pragma unroll
      for (int rt = 0; rt < 4; ++rt) {
        // A fragment: M = l16; i<8 -> K = kk*32+lh*8+i; i>=8 -> K = kk*32+16+lh*8+(i-8)
        FragBF a;
        const int rbase = (rt * 16 + l16) * STR1 + kk * 32 + lh * 8;
        const unsigned int* ap0 = (const unsigned int*)&lA[rbase];
        const unsigned int* ap1 = (const unsigned int*)&lA[rbase + 16];
        #pragma unroll
        for (int j = 0; j < 4; ++j) { a.u[j] = ap0[j]; a.u[4 + j] = ap1[j]; }
        c[rt] = __builtin_amdgcn_wmma_f32_16x16x32_bf16(false, a.bf, false, b.bf,
                                                        (short)0, c[rt], false, false);
      }
    }
    __syncthreads();
  }

  const int col = wave * 16 + l16;
  #pragma unroll
  for (int rt = 0; rt < 4; ++rt) {
    #pragma unroll
    for (int r = 0; r < 8; ++r) {
      int node = row0 + rt * 16 + lh * 8 + r;     // C: elem r -> M = r + lh*8
      if (node < n) feat1[(size_t)node * 128 + col] = c[rt][r];
    }
  }
}

// ============================================================
// GEMM2: feat2[N,16] = x2[N,128] @ W2[128,16]
// block: 256 thr = 8 waves; 128 rows, 16 cols, K=128
// ============================================================
__global__ __launch_bounds__(256) void gat_gemm2(const float* __restrict__ x2,
                                                 const float* __restrict__ W2,
                                                 float* __restrict__ feat2, int n) {
  __shared__ unsigned short lA[128 * STR1];  // 34.8 KB
  __shared__ unsigned short lW[16 * STR1];   //  4.4 KB
  const int t    = threadIdx.x;
  const int row0 = blockIdx.x * 128;
  const int wave = t >> 5;
  const int lane = t & 31;
  const int lh   = lane >> 4;
  const int l16  = lane & 15;

  for (int i = 0; i < 64; ++i) {
    int lin = i * 256 + t;
    int row = lin >> 7, k = lin & 127;
    int node = row0 + row;
    float v = (node < n) ? x2[(size_t)node * 128 + k] : 0.f;
    lA[row * STR1 + k] = f2bf(v);
  }
  for (int i = 0; i < 8; ++i) {
    int lin = i * 256 + t;               // lin = k*16 + col
    int k = lin >> 4, col = lin & 15;
    lW[col * STR1 + k] = f2bf(W2[lin]);
  }
  __syncthreads();

  v8f c = {0.f,0.f,0.f,0.f,0.f,0.f,0.f,0.f};
  #pragma unroll
  for (int kk = 0; kk < 4; ++kk) {
    FragBF a, b;
    const unsigned int* wp = (const unsigned int*)&lW[l16 * STR1 + kk * 32 + lh * 8];
    #pragma unroll
    for (int j = 0; j < 4; ++j) { b.u[j] = wp[j]; b.u[4 + j] = wp[8 + j]; }
    const int rbase = (wave * 16 + l16) * STR1 + kk * 32 + lh * 8;
    const unsigned int* ap0 = (const unsigned int*)&lA[rbase];
    const unsigned int* ap1 = (const unsigned int*)&lA[rbase + 16];
    #pragma unroll
    for (int j = 0; j < 4; ++j) { a.u[j] = ap0[j]; a.u[4 + j] = ap1[j]; }
    c = __builtin_amdgcn_wmma_f32_16x16x32_bf16(false, a.bf, false, b.bf,
                                                (short)0, c, false, false);
  }
  #pragma unroll
  for (int r = 0; r < 8; ++r) {
    int node = row0 + wave * 16 + lh * 8 + r;
    if (node < n) feat2[(size_t)node * 16 + l16] = c[r];
  }
}

// ============================================================
// Attention scores, edge softmax, aggregation
// ============================================================
__global__ void gat_fill(float* __restrict__ p, float v, int cnt) {
  int i = blockIdx.x * blockDim.x + threadIdx.x;
  if (i < cnt) p[i] = v;
}

__global__ void gat_scores1(const float* __restrict__ feat, const float* __restrict__ al,
                            const float* __restrict__ ar, float* __restrict__ el,
                            float* __restrict__ er, int n) {
  int t = blockIdx.x * blockDim.x + threadIdx.x;
  if (t >= n * 4) return;
  int hd = t & 3;
  const float* f = feat + (size_t)(t >> 2) * 128 + hd * 32;
  float sl = 0.f, sr = 0.f;
  for (int d = 0; d < 32; ++d) { float v = f[d]; sl += v * al[hd*32+d]; sr += v * ar[hd*32+d]; }
  el[t] = sl; er[t] = sr;
}

__global__ void gat_scores2(const float* __restrict__ feat, const float* __restrict__ al,
                            const float* __restrict__ ar, float* __restrict__ el,
                            float* __restrict__ er, int n) {
  int t = blockIdx.x * blockDim.x + threadIdx.x;
  if (t >= n) return;
  const float* f = feat + (size_t)t * 16;
  float sl = 0.f, sr = 0.f;
  for (int d = 0; d < 16; ++d) { float v = f[d]; sl += v * al[d]; sr += v * ar[d]; }
  el[t] = sl; er[t] = sr;
}

__device__ __forceinline__ float lrelu(float v) { return v > 0.f ? v : 0.2f * v; }

__global__ void gat_emax1(const int* __restrict__ src, const int* __restrict__ dst,
                          const float* __restrict__ el, const float* __restrict__ er,
                          float* __restrict__ emax, int e) {
  int t = blockIdx.x * blockDim.x + threadIdx.x;
  if (t >= e * 4) return;
  int ed = t >> 2, hd = t & 3;
  int s = src[ed], d = dst[ed];
  atomicMaxFloat(&emax[d*4+hd], lrelu(el[s*4+hd] + er[d*4+hd]));
}

__global__ void gat_eden1(const int* __restrict__ src, const int* __restrict__ dst,
                          const float* __restrict__ el, const float* __restrict__ er,
                          const float* __restrict__ emax, float* __restrict__ den, int e) {
  int t = blockIdx.x * blockDim.x + threadIdx.x;
  if (t >= e * 4) return;
  int ed = t >> 2, hd = t & 3;
  int s = src[ed], d = dst[ed];
  float v = lrelu(el[s*4+hd] + er[d*4+hd]);
  atomicAdd(&den[d*4+hd], __expf(v - emax[d*4+hd]));
}

__global__ void gat_aggr1(const int* __restrict__ src, const int* __restrict__ dst,
                          const float* __restrict__ el, const float* __restrict__ er,
                          const float* __restrict__ emax, const float* __restrict__ den,
                          const float* __restrict__ feat1, float* __restrict__ out1, int e) {
  int t = blockIdx.x * blockDim.x + threadIdx.x;   // e*32 threads, float4 each
  if (t >= e * 32) return;
  int ed = t >> 5, q = t & 31, hd = q >> 3;
  int s = src[ed], d = dst[ed];
  float v = lrelu(el[s*4+hd] + er[d*4+hd]);
  float a = __expf(v - emax[d*4+hd]) / den[d*4+hd];
  const float4 f = *(const float4*)&feat1[(size_t)s * 128 + q * 4];
  float* o = &out1[(size_t)d * 128 + q * 4];
  atomicAdd(o + 0, f.x * a); atomicAdd(o + 1, f.y * a);
  atomicAdd(o + 2, f.z * a); atomicAdd(o + 3, f.w * a);
}

__global__ void gat_elu(const float* __restrict__ out1, const float* __restrict__ b1,
                        float* __restrict__ x2, int n) {
  int t = blockIdx.x * blockDim.x + threadIdx.x;
  if (t >= n * 128) return;
  float v = out1[t] + b1[t & 127];
  x2[t] = v > 0.f ? v : __expf(v) - 1.0f;
}

__global__ void gat_emax2(const int* __restrict__ src, const int* __restrict__ dst,
                          const float* __restrict__ el, const float* __restrict__ er,
                          float* __restrict__ emax, int e) {
  int t = blockIdx.x * blockDim.x + threadIdx.x;
  if (t >= e) return;
  atomicMaxFloat(&emax[dst[t]], lrelu(el[src[t]] + er[dst[t]]));
}

__global__ void gat_eden2(const int* __restrict__ src, const int* __restrict__ dst,
                          const float* __restrict__ el, const float* __restrict__ er,
                          const float* __restrict__ emax, float* __restrict__ den, int e) {
  int t = blockIdx.x * blockDim.x + threadIdx.x;
  if (t >= e) return;
  int s = src[t], d = dst[t];
  float v = lrelu(el[s] + er[d]);
  atomicAdd(&den[d], __expf(v - emax[d]));
}

__global__ void gat_aggr2(const int* __restrict__ src, const int* __restrict__ dst,
                          const float* __restrict__ el, const float* __restrict__ er,
                          const float* __restrict__ emax, const float* __restrict__ den,
                          const float* __restrict__ feat2, float* __restrict__ out, int e) {
  int t = blockIdx.x * blockDim.x + threadIdx.x;   // e*4 threads, float4 each
  if (t >= e * 4) return;
  int ed = t >> 2, q = t & 3;
  int s = src[ed], d = dst[ed];
  float v = lrelu(el[s] + er[d]);
  float a = __expf(v - emax[d]) / den[d];
  const float4 f = *(const float4*)&feat2[(size_t)s * 16 + q * 4];
  float* o = &out[(size_t)d * 16 + q * 4];
  atomicAdd(o + 0, f.x * a); atomicAdd(o + 1, f.y * a);
  atomicAdd(o + 2, f.z * a); atomicAdd(o + 3, f.w * a);
}

__global__ void gat_bias(float* __restrict__ out, const float* __restrict__ b2, int n) {
  int t = blockIdx.x * blockDim.x + threadIdx.x;
  if (t >= n * 16) return;
  out[t] += b2[t & 15];
}

// ============================================================
extern "C" void kernel_launch(void* const* d_in, const int* in_sizes, int n_in,
                              void* d_out, int out_size, void* d_ws, size_t ws_size,
                              hipStream_t stream) {
  const float* h   = (const float*)d_in[0];
  const int*   src = (const int*)d_in[1];
  const int*   dst = (const int*)d_in[2];
  const float* W1  = (const float*)d_in[3];
  const float* al1 = (const float*)d_in[4];
  const float* ar1 = (const float*)d_in[5];
  const float* b1  = (const float*)d_in[6];
  const float* W2  = (const float*)d_in[7];
  const float* al2 = (const float*)d_in[8];
  const float* ar2 = (const float*)d_in[9];
  const float* b2  = (const float*)d_in[10];
  float* out = (float*)d_out;

  const int n = in_sizes[0] / 256;   // 50000
  const int e = in_sizes[1];         // 800000

  float* ws    = (float*)d_ws;
  float* feat1 = ws;                           // [n*128]  (reused as x2 after ELU)
  float* out1  = feat1 + (size_t)n * 128;      // [n*128]
  float* el1   = out1  + (size_t)n * 128;      // [n*4]
  float* er1   = el1   + (size_t)n * 4;
  float* emax1 = er1   + (size_t)n * 4;
  float* den1  = emax1 + (size_t)n * 4;
  float* feat2 = den1  + (size_t)n * 4;        // [n*16]
  float* el2   = feat2 + (size_t)n * 16;       // [n]
  float* er2   = el2   + n;
  float* emax2 = er2   + n;
  float* den2  = emax2 + n;

  const float NEG_INF = -__builtin_huge_valf();
  const int T = 256;
  auto G = [](int work, int thr) { return (work + thr - 1) / thr; };

  // init accumulators every call (workspace/d_out are poisoned)
  gat_fill<<<G(n * 128, T), T, 0, stream>>>(out1, 0.f, n * 128);
  gat_fill<<<G(n * 4, T),   T, 0, stream>>>(emax1, NEG_INF, n * 4);
  gat_fill<<<G(n * 4, T),   T, 0, stream>>>(den1, 0.f, n * 4);
  gat_fill<<<G(n, T),       T, 0, stream>>>(emax2, NEG_INF, n);
  gat_fill<<<G(n, T),       T, 0, stream>>>(den2, 0.f, n);
  gat_fill<<<G(n * 16, T),  T, 0, stream>>>(out, 0.f, n * 16);

  // layer 1
  gat_gemm1<<<G(n, 64), T, 0, stream>>>(h, W1, feat1, n);
  gat_scores1<<<G(n * 4, T), T, 0, stream>>>(feat1, al1, ar1, el1, er1, n);
  gat_emax1<<<G(e * 4, T), T, 0, stream>>>(src, dst, el1, er1, emax1, e);
  gat_eden1<<<G(e * 4, T), T, 0, stream>>>(src, dst, el1, er1, emax1, den1, e);
  gat_aggr1<<<G(e * 32, T), T, 0, stream>>>(src, dst, el1, er1, emax1, den1, feat1, out1, e);
  gat_elu<<<G(n * 128, T), T, 0, stream>>>(out1, b1, feat1, n);   // x2 -> feat1 buffer

  // layer 2
  gat_gemm2<<<G(n, 128), T, 0, stream>>>(feat1, W2, feat2, n);
  gat_scores2<<<G(n, T), T, 0, stream>>>(feat2, al2, ar2, el2, er2, n);
  gat_emax2<<<G(e, T), T, 0, stream>>>(src, dst, el2, er2, emax2, e);
  gat_eden2<<<G(e, T), T, 0, stream>>>(src, dst, el2, er2, emax2, den2, e);
  gat_aggr2<<<G(e * 4, T), T, 0, stream>>>(src, dst, el2, er2, emax2, den2, feat2, out, e);
  gat_bias<<<G(n * 16, T), T, 0, stream>>>(out, b2, n);
}
